// Attention_25142738551143
// MI455X (gfx1250) — matrix operand order repaired and checked by reading, not compiled, not run on hardware
//
#include <hip/hip_runtime.h>
#include <math.h>

typedef __bf16 bf16;
typedef __attribute__((ext_vector_type(4)))  bf16  v4bf;
typedef __attribute__((ext_vector_type(8)))  bf16  v8bf;
typedef __attribute__((ext_vector_type(16))) bf16  v16bf;
typedef __attribute__((ext_vector_type(8)))  float v8f;

#define NB   16          // batch
#define NL   2048        // sequence
#define NE   1024        // encoder dim
#define ND   1024        // decoder dim
#define MTILE 64         // rows of encoder per workgroup
#define AS_STRIDE (NE + 8)  // padded bf16 row stride: row step = 2064B -> conflict-free b128 LDS reads

union AFrag { v16bf v; v8bf h[2]; };

// ---------------------------------------------------------------------------
// Kernel 0a: t2[b,d] = decoder_hidden[b,:] @ W_dec[:,d] + b_dec[d] + b_enc[d]
// ---------------------------------------------------------------------------
__global__ void att2_kernel(const float* __restrict__ dec,
                            const float* __restrict__ Wdec,
                            const float* __restrict__ bdec,
                            const float* __restrict__ benc,
                            float* __restrict__ t2) {
    const int b = blockIdx.y;
    const int d = blockIdx.x * 256 + threadIdx.x;
    const float* x = dec + (size_t)b * ND;
    const float* w = Wdec + d;
    float acc = bdec[d] + benc[d];
#pragma unroll 4
    for (int e = 0; e < ND; ++e)
        acc = fmaf(x[e], w[(size_t)e * ND], acc);
    t2[(size_t)b * ND + d] = acc;
}

// ---------------------------------------------------------------------------
// Kernel 0b: W_encT[d][e] = (bf16) W_enc[e][d]  (LDS-tiled transpose+convert)
// ---------------------------------------------------------------------------
__global__ void transpose_cvt_kernel(const float* __restrict__ W,
                                     bf16* __restrict__ Wt) {
    __shared__ float tile[32][33];
    const int d0 = blockIdx.x * 32;
    const int e0 = blockIdx.y * 32;
    for (int i = threadIdx.y; i < 32; i += 8)
        tile[i][threadIdx.x] = W[(size_t)(e0 + i) * ND + d0 + threadIdx.x];
    __syncthreads();
    for (int i = threadIdx.y; i < 32; i += 8)
        Wt[(size_t)(d0 + i) * NE + e0 + threadIdx.x] = (bf16)tile[threadIdx.x][i];
}

// ---------------------------------------------------------------------------
// Kernel 1: fused score kernel, register-blocked over all 4 row-tiles.
// Each wave owns 8 N-tiles; every B fragment (one 32B load from L2-resident
// W_encT) feeds 4 WMMAs against 4 LDS A fragments -> 4x less global traffic.
// ---------------------------------------------------------------------------
__global__ void __launch_bounds__(256, 2)
attn_scores_kernel(const float* __restrict__ enc,
                   const bf16*  __restrict__ WencT,   // [ND][NE] bf16
                   const float* __restrict__ t2,      // [NB][ND]
                   const float* __restrict__ wfull,   // [ND]
                   const float* __restrict__ bfull,   // [1]
                   float* __restrict__ scoresOut) {   // [NB][NL]
    __shared__ __align__(16) bf16 As[MTILE * AS_STRIDE];
    __shared__ float scoreSh[MTILE];

    const int b  = blockIdx.y;
    const int l0 = blockIdx.x * MTILE;
    const int tid = threadIdx.x;

    // ---- stage A tile: 64 x 1024 fp32 -> bf16 in LDS (padded rows) ----
    const size_t encBase = ((size_t)b * NL + l0) * NE;
    for (int i = tid; i < MTILE * (NE / 4); i += 256) {
        const int row = i >> 8;          // NE/4 = 256 float4 per row
        const int c4  = i & 255;
        float4 f = *(const float4*)(enc + encBase + (size_t)row * NE + c4 * 4);
        v4bf p = { (bf16)f.x, (bf16)f.y, (bf16)f.z, (bf16)f.w };
        *(v4bf*)(&As[row * AS_STRIDE + c4 * 4]) = p;   // ds_store_b64
    }
    if (tid < MTILE) scoreSh[tid] = 0.0f;
    __syncthreads();

    const int wid  = tid >> 5;
    const int lane = tid & 31;
    const int hi   = lane >> 4;       // half-wave select
    const int lr   = lane & 15;
    const int nt0  = wid * 8;         // 8 waves x 8 N-tiles = 64

    const bf16* a0p = &As[( 0 + lr) * AS_STRIDE + hi * 8];
    const bf16* a1p = &As[(16 + lr) * AS_STRIDE + hi * 8];
    const bf16* a2p = &As[(32 + lr) * AS_STRIDE + hi * 8];
    const bf16* a3p = &As[(48 + lr) * AS_STRIDE + hi * 8];

    float s[4][8];
#pragma unroll
    for (int r = 0; r < 4; ++r)
#pragma unroll
        for (int j = 0; j < 8; ++j) s[r][j] = 0.0f;

    for (int nt = nt0; nt < nt0 + 8; ++nt) {
        v8f acc0 = {}, acc1 = {}, acc2 = {}, acc3 = {};
        const bf16* bRow = WencT + (size_t)(nt * 16 + lr) * NE + hi * 8;
#pragma unroll 2
        for (int k0 = 0; k0 < NE; k0 += 32) {
            // B 32x16 bf16 fragment: lanes 0-15 K {0-7,16-23}, 16-31 {+8}
            AFrag bf;
            bf.h[0] = *(const v8bf*)(bRow + k0);
            bf.h[1] = *(const v8bf*)(bRow + k0 + 16);
            v16bf bv = bf.v;
            // A 16x32 bf16 fragments: lanes 0-15 K {0-7,16-23}, 16-31 {+8}
            AFrag a0, a1, a2, a3;
            a0.h[0] = *(const v8bf*)(a0p + k0);
            a0.h[1] = *(const v8bf*)(a0p + k0 + 16);
            a1.h[0] = *(const v8bf*)(a1p + k0);
            a1.h[1] = *(const v8bf*)(a1p + k0 + 16);
            a2.h[0] = *(const v8bf*)(a2p + k0);
            a2.h[1] = *(const v8bf*)(a2p + k0 + 16);
            a3.h[0] = *(const v8bf*)(a3p + k0);
            a3.h[1] = *(const v8bf*)(a3p + k0 + 16);
            acc0 = __builtin_amdgcn_wmma_f32_16x16x32_bf16(
                false, a0.v, false, bv, (short)0, acc0, false, false);
            acc1 = __builtin_amdgcn_wmma_f32_16x16x32_bf16(
                false, a1.v, false, bv, (short)0, acc1, false, false);
            acc2 = __builtin_amdgcn_wmma_f32_16x16x32_bf16(
                false, a2.v, false, bv, (short)0, acc2, false, false);
            acc3 = __builtin_amdgcn_wmma_f32_16x16x32_bf16(
                false, a3.v, false, bv, (short)0, acc3, false, false);
        }
        // C layout: element j of accR = (row R*16 + hi*8 + j, col nt*16 + lr)
        const int col = nt * 16 + lr;
        const float t2v = t2[(size_t)b * ND + col];
        const float wf  = wfull[col];
#pragma unroll
        for (int j = 0; j < 8; ++j) {
            float x0 = acc0[j] + t2v; x0 = x0 > 0.0f ? x0 : 0.0f;
            float x1 = acc1[j] + t2v; x1 = x1 > 0.0f ? x1 : 0.0f;
            float x2 = acc2[j] + t2v; x2 = x2 > 0.0f ? x2 : 0.0f;
            float x3 = acc3[j] + t2v; x3 = x3 > 0.0f ? x3 : 0.0f;
            s[0][j] = fmaf(x0, wf, s[0][j]);
            s[1][j] = fmaf(x1, wf, s[1][j]);
            s[2][j] = fmaf(x2, wf, s[2][j]);
            s[3][j] = fmaf(x3, wf, s[3][j]);
        }
    }

    // reduce the 16 column-lanes of each half-wave, then combine waves via LDS
#pragma unroll
    for (int r = 0; r < 4; ++r) {
#pragma unroll
        for (int j = 0; j < 8; ++j) {
            float v = s[r][j];
            v += __shfl_xor(v, 1, 32);
            v += __shfl_xor(v, 2, 32);
            v += __shfl_xor(v, 4, 32);
            v += __shfl_xor(v, 8, 32);
            if (lr == 0)
                atomicAdd(&scoreSh[r * 16 + hi * 8 + j], v);
        }
    }
    __syncthreads();

    if (tid < MTILE)
        scoresOut[(size_t)b * NL + l0 + tid] = scoreSh[tid] + bfull[0];
}

// ---------------------------------------------------------------------------
// Kernel 2: in-place softmax over L per batch (one workgroup per b)
// ---------------------------------------------------------------------------
__global__ void softmax_kernel(float* __restrict__ alpha) {
    __shared__ float redm[8];
    __shared__ float reds[8];
    const int b = blockIdx.x;
    float* p = alpha + (size_t)b * NL;
    const int tid = threadIdx.x;

    float vals[8];
    float mx = -INFINITY;
#pragma unroll
    for (int i = 0; i < 8; ++i) {
        vals[i] = p[tid + i * 256];
        mx = fmaxf(mx, vals[i]);
    }
    for (int off = 1; off < 32; off <<= 1) mx = fmaxf(mx, __shfl_xor(mx, off, 32));
    if ((tid & 31) == 0) redm[tid >> 5] = mx;
    __syncthreads();
    float m = redm[0];
#pragma unroll
    for (int i = 1; i < 8; ++i) m = fmaxf(m, redm[i]);

    float sum = 0.0f;
#pragma unroll
    for (int i = 0; i < 8; ++i) {
        vals[i] = expf(vals[i] - m);
        sum += vals[i];
    }
    for (int off = 1; off < 32; off <<= 1) sum += __shfl_xor(sum, off, 32);
    if ((tid & 31) == 0) reds[tid >> 5] = sum;
    __syncthreads();
    float tot = 0.0f;
#pragma unroll
    for (int i = 0; i < 8; ++i) tot += reds[i];

    const float inv = 1.0f / tot;
#pragma unroll
    for (int i = 0; i < 8; ++i)
        p[tid + i * 256] = vals[i] * inv;
}

// ---------------------------------------------------------------------------
// Kernel 3: context[b,e] = sum_l alpha[b,l] * enc[b,l,e]  (bandwidth bound)
// ---------------------------------------------------------------------------
__global__ void context_kernel(const float* __restrict__ enc,
                               const float* __restrict__ alpha,
                               float* __restrict__ ctx) {
    __shared__ float aSh[NL];
    const int b = blockIdx.y;
    const int e = blockIdx.x * 256 + threadIdx.x;
    for (int i = threadIdx.x; i < NL; i += 256)
        aSh[i] = alpha[(size_t)b * NL + i];
    __syncthreads();
    const float* ep = enc + (size_t)b * NL * NE + e;
    float acc = 0.0f;
#pragma unroll 4
    for (int l = 0; l < NL; ++l)
        acc = fmaf(aSh[l], ep[(size_t)l * NE], acc);
    ctx[(size_t)b * NE + e] = acc;
}

// ---------------------------------------------------------------------------
extern "C" void kernel_launch(void* const* d_in, const int* in_sizes, int n_in,
                              void* d_out, int out_size, void* d_ws, size_t ws_size,
                              hipStream_t stream) {
    (void)in_sizes; (void)n_in; (void)out_size; (void)ws_size;
    const float* enc   = (const float*)d_in[0];  // [16,2048,1024]
    const float* dec   = (const float*)d_in[1];  // [16,1024]
    const float* Wenc  = (const float*)d_in[2];  // [1024,1024]
    const float* benc  = (const float*)d_in[3];  // [1024]
    const float* Wdec  = (const float*)d_in[4];  // [1024,1024]
    const float* bdec  = (const float*)d_in[5];  // [1024]
    const float* wfull = (const float*)d_in[6];  // [1024]
    const float* bfull = (const float*)d_in[7];  // [1]

    float* ctx   = (float*)d_out;                 // [16,1024]
    float* alpha = (float*)d_out + NB * NE;       // [16,2048] (holds raw scores first)

    // workspace: t2 (64 KB f32) then W_encT (2 MB bf16)
    float* t2    = (float*)d_ws;
    bf16*  WencT = (bf16*)((char*)d_ws + (size_t)NB * ND * sizeof(float));

    att2_kernel<<<dim3(ND / 256, NB), 256, 0, stream>>>(dec, Wdec, bdec, benc, t2);
    transpose_cvt_kernel<<<dim3(ND / 32, NE / 32), dim3(32, 8), 0, stream>>>(Wenc, WencT);
    attn_scores_kernel<<<dim3(NL / MTILE, NB), 256, 0, stream>>>(
        enc, WencT, t2, wfull, bfull, alpha);
    softmax_kernel<<<NB, 256, 0, stream>>>(alpha);
    context_kernel<<<dim3(NE / 256, NB), 256, 0, stream>>>(enc, alpha, ctx);
}
